// SwinTransformerBlock_1228360647057
// MI455X (gfx1250) — matrix operand order repaired and checked by reading, not compiled, not run on hardware
//
#include <hip/hip_runtime.h>
#include <cstdint>
#include <cstddef>

// ---------------- types ----------------
typedef _Float16 h16;
typedef _Float16 v16h __attribute__((ext_vector_type(16)));
typedef _Float16 v8h  __attribute__((ext_vector_type(8)));
typedef float    v8f  __attribute__((ext_vector_type(8)));

// ---------------- problem dims ----------------
#define BATCH 4
#define CDIM  256
#define NHEAD 8
#define DHEAD 32
#define NWIN  48
#define PWIN  192
#define GRID  96
#define FFDIM 1024
#define MROWS (BATCH * NWIN * PWIN)      // 36864
#define SCALE_QK 0.0625f                 // 1/sqrt(256)

// ---------------- workspace layout (bytes) ----------------
#define OFF_A    0                                   // fp32 residual a   [M,256]
#define OFF_YH   (OFF_A  + (size_t)MROWS*CDIM*4)     // f16 ln1 out / ln2 out (reused)
#define OFF_Q    (OFF_YH + (size_t)MROWS*CDIM*2)     // f16 [BW,NH,P,Dh]
#define OFF_K    (OFF_Q  + (size_t)MROWS*CDIM*2)     // f16 [BW,NH,P,Dh]
#define OFF_VT   (OFF_K  + (size_t)MROWS*CDIM*2)     // f16 [BW,NH,Dh,P]
#define OFF_SA   (OFF_VT + (size_t)MROWS*CDIM*2)     // f16 [M,256]
#define OFF_FF   OFF_Q                               // f16 [M,1024]  (aliases Q..SA, safe by stream order)
#define OFF_WQT  (OFF_SA + (size_t)MROWS*CDIM*2)
#define OFF_WKT  (OFF_WQT + (size_t)CDIM*CDIM*2)
#define OFF_WVT  (OFF_WKT + (size_t)CDIM*CDIM*2)
#define OFF_W1T  (OFF_WVT + (size_t)CDIM*CDIM*2)     // [1024,256] f16
#define OFF_W2T  (OFF_W1T + (size_t)CDIM*FFDIM*2)    // [256,1024] f16

// ---------------- WMMA helpers ----------------
// A fragment, 16x32 f16, row-major src with leading dim ld (elements).
// Lane L: row m = L&15; contiguous K halves {hg*16..+15}, hg = L>>4.
__device__ __forceinline__ v16h frag_a(const h16* base, int ld) {
    const int lane = threadIdx.x & 31;
    const h16* r = base + (size_t)(lane & 15) * ld + ((lane >> 4) << 4);
    return *(const v16h*)(r);
}

// B fragment from B-transposed storage Bt[N][K] (row-major, ld elements).
// Lane L: col n = L&15; contiguous K halves {hg*16..+15}.
__device__ __forceinline__ v16h frag_b(const h16* bt, int ld) {
    const int lane = threadIdx.x & 31;
    return *(const v16h*)(bt + (size_t)(lane & 15) * ld + ((lane >> 4) << 4));
}

__device__ __forceinline__ v8f wmma32(v16h a, v16h b, v8f c) {
    return __builtin_amdgcn_wmma_f32_16x16x32_f16(false, a, false, b, (short)0, c,
                                                  false, false);
}

// roll(+16,+3) -> partition and join -> roll(-16,-3) cancel: both gather and
// scatter use the same (row -> b,gy,gx) map.
__device__ __forceinline__ void row_to_grid(int row, int& b, int& gy, int& gx) {
    b = row / (NWIN * PWIN);
    int rem = row - b * (NWIN * PWIN);
    int w = rem / PWIN, p = rem - (rem / PWIN) * PWIN;
    int vw = w / 3, hw = w - vw * 3;
    int vp = p >> 5, hp = p & 31;
    gy = (vp * 16 + vw + 80) % GRID;   // (i-16) mod 96
    gx = (hp * 3 + hw + 93) % GRID;    // (j-3)  mod 96
}

// ---------------- k0: weight transpose fp32[R][C] -> f16[C][R] ----------------
__global__ __launch_bounds__(256) void wtrans_kernel(const float* __restrict__ src,
                                                     h16* __restrict__ dst,
                                                     int R, int C) {
    int idx = blockIdx.x * 256 + threadIdx.x;
    if (idx >= R * C) return;
    int r = idx / C, c = idx - r * C;
    dst[(size_t)c * R + r] = (h16)src[(size_t)r * C + c];
}

// ---------------- k1: gather + LN1 -> yh (f16) ----------------
__global__ __launch_bounds__(256) void ln1_kernel(const float* __restrict__ emb,
                                                  const float* __restrict__ g,
                                                  const float* __restrict__ bb,
                                                  h16* __restrict__ yh) {
    int row  = blockIdx.x * 8 + (threadIdx.x >> 5);
    int lane = threadIdx.x & 31;
    int b, gy, gx; row_to_grid(row, b, gy, gx);
    const float* src = emb + (((size_t)b * GRID + gy) * GRID + gx) * CDIM;
    float x[8], s = 0.f;
#pragma unroll
    for (int t = 0; t < 8; ++t) { x[t] = src[lane + 32 * t]; s += x[t]; }
#pragma unroll
    for (int off = 16; off >= 1; off >>= 1) s += __shfl_xor(s, off);
    float mean = s * (1.0f / CDIM), v = 0.f;
#pragma unroll
    for (int t = 0; t < 8; ++t) { float d = x[t] - mean; v += d * d; }
#pragma unroll
    for (int off = 16; off >= 1; off >>= 1) v += __shfl_xor(v, off);
    float rs = rsqrtf(v * (1.0f / CDIM) + 1e-5f);
    h16* dst = yh + (size_t)row * CDIM;
#pragma unroll
    for (int t = 0; t < 8; ++t) {
        int c = lane + 32 * t;
        dst[c] = (h16)((x[t] - mean) * rs * g[c] + bb[c]);
    }
}

// ---------------- k2: fused QKV GEMM ----------------
// Block = 4 waves; wave w owns the 64-col strip n0 = w*64 of a 16-row tile.
// All 12 B-fragments of a k-step are loaded into distinct registers before the
// WMMAs so the scheduler can issue partial s_wait_loadcnt and overlap.
__global__ __launch_bounds__(128) void qkv_kernel(const h16* __restrict__ yh,
                                                 const h16* __restrict__ wqt,
                                                 const h16* __restrict__ wkt,
                                                 const h16* __restrict__ wvt,
                                                 const float* __restrict__ bq,
                                                 const float* __restrict__ bk,
                                                 const float* __restrict__ bv,
                                                 h16* __restrict__ q,
                                                 h16* __restrict__ k,
                                                 h16* __restrict__ vt) {
    const int mt = blockIdx.x;           // 16-row tile, 192%16==0 -> one window
    const int n0 = (threadIdx.x >> 5) * 64;
    const int lane = threadIdx.x & 31, nn = lane & 15, hg = lane >> 4;
    const h16* arow = yh + (size_t)mt * 16 * CDIM;
    v8f aq[4], ak[4], av[4];
#pragma unroll
    for (int j = 0; j < 4; ++j) { aq[j] = (v8f)0.f; ak[j] = (v8f)0.f; av[j] = (v8f)0.f; }
    for (int kk = 0; kk < CDIM; kk += 32) {
        v16h A = frag_a(arow + kk, CDIM);
        v16h Bq[4], Bk[4], Bv[4];
#pragma unroll
        for (int j = 0; j < 4; ++j) {
            size_t nb = (size_t)(n0 + j * 16) * CDIM + kk;
            Bq[j] = frag_b(wqt + nb, CDIM);
            Bk[j] = frag_b(wkt + nb, CDIM);
            Bv[j] = frag_b(wvt + nb, CDIM);
        }
#pragma unroll
        for (int j = 0; j < 4; ++j) {
            aq[j] = wmma32(A, Bq[j], aq[j]);
            ak[j] = wmma32(A, Bk[j], ak[j]);
            av[j] = wmma32(A, Bv[j], av[j]);
        }
    }
    const int bw = (mt * 16) / PWIN, pb = (mt * 16) % PWIN;
#pragma unroll
    for (int j = 0; j < 4; ++j) {
        int c = n0 + j * 16 + nn, hh = c >> 5, d = c & 31;
        float bqc = bq[c], bkc = bk[c], bvc = bv[c];
#pragma unroll
        for (int r = 0; r < 8; ++r) {
            int p = pb + r + 8 * hg;
            size_t head = (size_t)bw * NHEAD + hh;
            q [(head * PWIN + p) * DHEAD + d] = (h16)(aq[j][r] + bqc);
            k [(head * PWIN + p) * DHEAD + d] = (h16)(ak[j][r] + bkc);
            vt[(head * DHEAD + d) * PWIN + p] = (h16)(av[j][r] + bvc);
        }
    }
}

// ---------------- k3: attention, one block per (window,head), 12 waves --------
__global__ __launch_bounds__(384) void attn_kernel(const h16* __restrict__ q,
                                                   const h16* __restrict__ k,
                                                   const h16* __restrict__ vt,
                                                   const float* __restrict__ pos_bias,
                                                   h16* __restrict__ sa) {
    const int bwh = blockIdx.x;            // 0..1535
    const int h = bwh & 7, bw = bwh >> 3;
    const int wv = threadIdx.x >> 5, lane = threadIdx.x & 31;
    const int nn = lane & 15, hg = lane >> 4;
    const int m0 = wv * 16;
    const h16* qb = q  + (size_t)bwh * PWIN * DHEAD;
    const h16* kb = k  + (size_t)bwh * PWIN * DHEAD;
    const h16* vb = vt + (size_t)bwh * DHEAD * PWIN;
    const float* pb = pos_bias + (size_t)h * PWIN * PWIN;

    __shared__ __align__(32) h16 lattn[12][16][PWIN];

    // logits strip [16 x 192]: Dh=32 -> a single WMMA per 16x16 tile
    v16h Aq = frag_a(qb + (size_t)m0 * DHEAD, DHEAD);
    v8f acc[12];
#pragma unroll
    for (int j = 0; j < 12; ++j) {
        acc[j] = (v8f)0.f;
        acc[j] = wmma32(Aq, frag_b(kb + (size_t)(j * 16) * DHEAD, DHEAD), acc[j]);
    }
    // scale + positional bias; row max (row = r+8*hg lives in one 16-lane half)
    float mx[8];
#pragma unroll
    for (int r = 0; r < 8; ++r) mx[r] = -3.0e38f;
#pragma unroll
    for (int j = 0; j < 12; ++j)
#pragma unroll
        for (int r = 0; r < 8; ++r) {
            float val = acc[j][r] * SCALE_QK +
                        pb[(size_t)(m0 + r + 8 * hg) * PWIN + j * 16 + nn];
            acc[j][r] = val;
            mx[r] = fmaxf(mx[r], val);
        }
#pragma unroll
    for (int off = 8; off >= 1; off >>= 1)
#pragma unroll
        for (int r = 0; r < 8; ++r) mx[r] = fmaxf(mx[r], __shfl_xor(mx[r], off));
    float sm[8];
#pragma unroll
    for (int r = 0; r < 8; ++r) sm[r] = 0.f;
#pragma unroll
    for (int j = 0; j < 12; ++j)
#pragma unroll
        for (int r = 0; r < 8; ++r) {
            float e = __expf(acc[j][r] - mx[r]);
            acc[j][r] = e; sm[r] += e;
        }
#pragma unroll
    for (int off = 8; off >= 1; off >>= 1)
#pragma unroll
        for (int r = 0; r < 8; ++r) sm[r] += __shfl_xor(sm[r], off);
#pragma unroll
    for (int r = 0; r < 8; ++r) sm[r] = 1.0f / sm[r];

    // stage normalized probs to LDS so they can be re-read in A-fragment layout
#pragma unroll
    for (int j = 0; j < 12; ++j)
#pragma unroll
        for (int r = 0; r < 8; ++r)
            lattn[wv][r + 8 * hg][j * 16 + nn] = (h16)(acc[j][r] * sm[r]);
    __syncthreads();

    // sa strip = attn[16x192] @ V[192x32]
    v8f o[2];
    o[0] = (v8f)0.f; o[1] = (v8f)0.f;
    const h16* abase = &lattn[wv][0][0];
    for (int kk = 0; kk < PWIN; kk += 32) {
        v16h Af = frag_a(abase + kk, PWIN);
        v16h B0 = frag_b(vb + kk, PWIN);
        v16h B1 = frag_b(vb + (size_t)16 * PWIN + kk, PWIN);
        o[0] = wmma32(Af, B0, o[0]);
        o[1] = wmma32(Af, B1, o[1]);
    }
#pragma unroll
    for (int j = 0; j < 2; ++j)
#pragma unroll
        for (int r = 0; r < 8; ++r) {
            int p = m0 + r + 8 * hg;
            int c = h * DHEAD + j * 16 + nn;
            sa[((size_t)bw * PWIN + p) * CDIM + c] = (h16)o[j][r];
        }
}

// ---------------- k4: residual a = gather(emb)+sa; LN2 -> h ----------------
__global__ __launch_bounds__(256) void ln2_kernel(const float* __restrict__ emb,
                                                  const h16* __restrict__ sa,
                                                  const float* __restrict__ g,
                                                  const float* __restrict__ bb,
                                                  float* __restrict__ a,
                                                  h16* __restrict__ hh) {
    int row  = blockIdx.x * 8 + (threadIdx.x >> 5);
    int lane = threadIdx.x & 31;
    int b, gy, gx; row_to_grid(row, b, gy, gx);
    const float* src = emb + (((size_t)b * GRID + gy) * GRID + gx) * CDIM;
    const h16* sr = sa + (size_t)row * CDIM;
    float x[8], s = 0.f;
#pragma unroll
    for (int t = 0; t < 8; ++t) {
        int c = lane + 32 * t;
        x[t] = src[c] + (float)sr[c];
        s += x[t];
    }
#pragma unroll
    for (int off = 16; off >= 1; off >>= 1) s += __shfl_xor(s, off);
    float mean = s * (1.0f / CDIM), v = 0.f;
#pragma unroll
    for (int t = 0; t < 8; ++t) { float d = x[t] - mean; v += d * d; }
#pragma unroll
    for (int off = 16; off >= 1; off >>= 1) v += __shfl_xor(v, off);
    float rs = rsqrtf(v * (1.0f / CDIM) + 1e-5f);
    float* adst = a + (size_t)row * CDIM;
    h16*   hdst = hh + (size_t)row * CDIM;
#pragma unroll
    for (int t = 0; t < 8; ++t) {
        int c = lane + 32 * t;
        adst[c] = x[t];
        hdst[c] = (h16)((x[t] - mean) * rs * g[c] + bb[c]);
    }
}

// ---------------- k5: FFN1 (256->1024) + exact GELU ----------------
// 32-row M tile per wave: each B-fragment feeds 2 WMMAs. Block = 8 waves
// covering 8 of 16 N-strips; grid.y = 2.
__global__ __launch_bounds__(256) void ffn1_kernel(const h16* __restrict__ hh,
                                                  const h16* __restrict__ w1t,
                                                  const float* __restrict__ b1,
                                                  h16* __restrict__ ff) {
    const int mt = blockIdx.x;
    const int n0 = (blockIdx.y * 8 + (threadIdx.x >> 5)) * 64;
    const int lane = threadIdx.x & 31, nn = lane & 15, hg = lane >> 4;
    const h16* arow = hh + (size_t)mt * 32 * CDIM;
    v8f acc[2][4];
#pragma unroll
    for (int mi = 0; mi < 2; ++mi)
#pragma unroll
        for (int j = 0; j < 4; ++j) acc[mi][j] = (v8f)0.f;
    for (int kk = 0; kk < CDIM; kk += 32) {
        v16h A0 = frag_a(arow + kk, CDIM);
        v16h A1 = frag_a(arow + (size_t)16 * CDIM + kk, CDIM);
        v16h Bf[4];
#pragma unroll
        for (int j = 0; j < 4; ++j)
            Bf[j] = frag_b(w1t + (size_t)(n0 + j * 16) * CDIM + kk, CDIM);
#pragma unroll
        for (int j = 0; j < 4; ++j) {
            acc[0][j] = wmma32(A0, Bf[j], acc[0][j]);
            acc[1][j] = wmma32(A1, Bf[j], acc[1][j]);
        }
    }
#pragma unroll
    for (int j = 0; j < 4; ++j) {
        int c = n0 + j * 16 + nn;
        float bc = b1[c];
#pragma unroll
        for (int mi = 0; mi < 2; ++mi)
#pragma unroll
            for (int r = 0; r < 8; ++r) {
                int row = mt * 32 + mi * 16 + r + 8 * hg;
                float x = acc[mi][j][r] + bc;
                float gl = 0.5f * x * (1.0f + erff(x * 0.70710678118654752f));
                ff[(size_t)row * FFDIM + c] = (h16)gl;
            }
    }
}

// ---------------- k6: FFN2 (1024->256) + residual + scatter ----------------
// 32-row M tile per wave; block = 4 waves covering all 4 N-strips.
__global__ __launch_bounds__(128) void ffn2_kernel(const h16* __restrict__ ff,
                                                  const h16* __restrict__ w2t,
                                                  const float* __restrict__ b2,
                                                  const float* __restrict__ a,
                                                  float* __restrict__ out) {
    const int mt = blockIdx.x;
    const int n0 = (threadIdx.x >> 5) * 64;
    const int lane = threadIdx.x & 31, nn = lane & 15, hg = lane >> 4;
    const h16* arow = ff + (size_t)mt * 32 * FFDIM;
    v8f acc[2][4];
#pragma unroll
    for (int mi = 0; mi < 2; ++mi)
#pragma unroll
        for (int j = 0; j < 4; ++j) acc[mi][j] = (v8f)0.f;
    for (int kk = 0; kk < FFDIM; kk += 32) {
        // prefetch next k-chunk of the A rows (gfx1250 global_prefetch path)
        if (kk + 32 < FFDIM) {
            __builtin_prefetch(arow + kk + 32, 0, 1);
            __builtin_prefetch(arow + (size_t)16 * FFDIM + kk + 32, 0, 1);
        }
        v16h A0 = frag_a(arow + kk, FFDIM);
        v16h A1 = frag_a(arow + (size_t)16 * FFDIM + kk, FFDIM);
        v16h Bf[4];
#pragma unroll
        for (int j = 0; j < 4; ++j)
            Bf[j] = frag_b(w2t + (size_t)(n0 + j * 16) * FFDIM + kk, FFDIM);
#pragma unroll
        for (int j = 0; j < 4; ++j) {
            acc[0][j] = wmma32(A0, Bf[j], acc[0][j]);
            acc[1][j] = wmma32(A1, Bf[j], acc[1][j]);
        }
    }
#pragma unroll
    for (int j = 0; j < 4; ++j) {
        int c = n0 + j * 16 + nn;
        float bc = b2[c];
#pragma unroll
        for (int mi = 0; mi < 2; ++mi)
#pragma unroll
            for (int r = 0; r < 8; ++r) {
                int row = mt * 32 + mi * 16 + r + 8 * hg;
                float val = acc[mi][j][r] + bc + a[(size_t)row * CDIM + c];
                int b, gy, gx; row_to_grid(row, b, gy, gx);
                out[(((size_t)b * GRID + gy) * GRID + gx) * CDIM + c] = val;
            }
    }
}

// ---------------- host launcher ----------------
extern "C" void kernel_launch(void* const* d_in, const int* in_sizes, int n_in,
                              void* d_out, int out_size, void* d_ws, size_t ws_size,
                              hipStream_t stream) {
    const float* emb   = (const float*)d_in[0];
    const float* ln1g  = (const float*)d_in[1];
    const float* ln1b  = (const float*)d_in[2];
    const float* wq    = (const float*)d_in[3];
    const float* bq    = (const float*)d_in[4];
    const float* wk    = (const float*)d_in[5];
    const float* bk    = (const float*)d_in[6];
    const float* wv    = (const float*)d_in[7];
    const float* bv    = (const float*)d_in[8];
    const float* posb  = (const float*)d_in[9];
    const float* ln2g  = (const float*)d_in[10];
    const float* ln2b  = (const float*)d_in[11];
    const float* w1    = (const float*)d_in[12];
    const float* b1    = (const float*)d_in[13];
    const float* w2    = (const float*)d_in[14];
    const float* b2    = (const float*)d_in[15];

    char* ws = (char*)d_ws;
    float* A   = (float*)(ws + OFF_A);
    h16*   YH  = (h16*)(ws + OFF_YH);   // also serves as LN2 output h
    h16*   Q   = (h16*)(ws + OFF_Q);
    h16*   K   = (h16*)(ws + OFF_K);
    h16*   VT  = (h16*)(ws + OFF_VT);
    h16*   SA  = (h16*)(ws + OFF_SA);
    h16*   FF  = (h16*)(ws + OFF_FF);   // aliases Q..SA after attention done
    h16*   WQT = (h16*)(ws + OFF_WQT);
    h16*   WKT = (h16*)(ws + OFF_WKT);
    h16*   WVT = (h16*)(ws + OFF_WVT);
    h16*   W1T = (h16*)(ws + OFF_W1T);
    h16*   W2T = (h16*)(ws + OFF_W2T);

    // k0: weights -> f16 transposed [N][K]
    wtrans_kernel<<<(CDIM * CDIM + 255) / 256, 256, 0, stream>>>(wq, WQT, CDIM, CDIM);
    wtrans_kernel<<<(CDIM * CDIM + 255) / 256, 256, 0, stream>>>(wk, WKT, CDIM, CDIM);
    wtrans_kernel<<<(CDIM * CDIM + 255) / 256, 256, 0, stream>>>(wv, WVT, CDIM, CDIM);
    wtrans_kernel<<<(CDIM * FFDIM + 255) / 256, 256, 0, stream>>>(w1, W1T, CDIM, FFDIM);
    wtrans_kernel<<<(CDIM * FFDIM + 255) / 256, 256, 0, stream>>>(w2, W2T, FFDIM, CDIM);

    // k1: gather + LN1
    ln1_kernel<<<MROWS / 8, 256, 0, stream>>>(emb, ln1g, ln1b, YH);

    // k2: QKV projection (M=36864, N=256 each, fused; 4 waves/block)
    qkv_kernel<<<MROWS / 16, 128, 0, stream>>>(
        YH, WQT, WKT, WVT, bq, bk, bv, Q, K, VT);

    // k3: windowed attention, one block per (window, head)
    attn_kernel<<<BATCH * NWIN * NHEAD, 384, 0, stream>>>(Q, K, VT, posb, SA);

    // k4: residual + LN2 (h reuses YH buffer)
    ln2_kernel<<<MROWS / 8, 256, 0, stream>>>(emb, SA, ln2g, ln2b, A, YH);

    // k5: FFN1 + GELU  (FF aliases Q..SA region, all consumed by now)
    ffn1_kernel<<<dim3(MROWS / 32, 2), 256, 0, stream>>>(YH, W1T, b1, FF);

    // k6: FFN2 + residual + scatter (join + inverse roll fused)
    ffn2_kernel<<<MROWS / 32, 128, 0, stream>>>(FF, W2T, b2, A, (float*)d_out);
}
